// CHARM_10677288698622
// MI455X (gfx1250) — matrix operand order repaired and checked by reading, not compiled, not run on hardware
//
#include <hip/hip_runtime.h>

// ---------------------------------------------------------------------------
// GNN message-passing network on MI455X (gfx1250).
// GEMMs via V_WMMA_F32_16X16X32_BF16 (bf16 operands, f32 accumulate).
// Edge-gather via GLOBAL_LOAD_ASYNC_TO_LDS_B128 (ASYNCcnt path).
// ---------------------------------------------------------------------------

typedef __bf16 bf16;
typedef __attribute__((ext_vector_type(16))) __bf16 v16bf;
typedef __attribute__((ext_vector_type(8)))  __bf16 v8bf;
typedef __attribute__((ext_vector_type(8)))  float  v8f;

#define N_NODES  50000
#define N_EDGES  400000
#define NODE_IN  128
#define EDGE_DIM 32
#define HIDDEN   256
#define MSG_K    (2*HIDDEN + EDGE_DIM)   // 544 = 17 * 32

// --- A-fragment (16x32 bf16) from an LDS row, per CDNA5 WMMA layout:
// lane = m + 16*khalf; elements 0..7 -> K = k0 + 8*khalf + i
//                      elements 8..15 -> K = k0 + 16 + 8*khalf + (i-8)
__device__ __forceinline__ v16bf load_a_tile(const bf16* rowPtr, int k0, int khalf) {
  v8bf lo = *(const v8bf*)(rowPtr + k0 + 8*khalf);        // ds_load_b128
  v8bf hi = *(const v8bf*)(rowPtr + k0 + 16 + 8*khalf);   // ds_load_b128
  v16bf a;
#pragma unroll
  for (int i = 0; i < 8; ++i) { a[i] = lo[i]; a[8+i] = hi[i]; }
  return a;
}

// --- B-fragment from weights pre-packed so each lane's 16 bf16 are contiguous.
__device__ __forceinline__ v16bf load_b_tile(const bf16* Bp, int KT, int nt, int kt, int lane) {
  return *(const v16bf*)(Bp + (((size_t)nt*KT + kt)*32 + lane)*16);
}

__device__ __forceinline__ v8f wmma_bf16(v16bf a, v16bf b, v8f c) {
  return __builtin_amdgcn_wmma_f32_16x16x32_bf16(false, a, false, b, (short)0, c, false, false);
}

// --- CDNA5 async copy: 16B global -> LDS, tracked by ASYNCcnt.
__device__ __forceinline__ void async_copy_b128(unsigned lds_off, const void* gptr) {
  unsigned long long g = (unsigned long long)(uintptr_t)gptr;
  asm volatile("global_load_async_to_lds_b128 %0, %1, off"
               :: "v"(lds_off), "v"(g) : "memory");
}
__device__ __forceinline__ void async_wait0() {
  asm volatile("s_wait_asynccnt 0x0" ::: "memory");
}
__device__ __forceinline__ unsigned lds_offset_of(const void* p) {
  // flat shared-aperture address: low 32 bits are the LDS byte offset
  return (unsigned)(uintptr_t)p;
}

// ---------------------------------------------------------------------------
// Weight pack: W[K,N] f32 row-major -> bf16 tiles in WMMA-B lane order.
// ---------------------------------------------------------------------------
__global__ void pack_w_kernel(const float* __restrict__ W, bf16* __restrict__ P,
                              int K, int N) {
  int p = blockIdx.x * 256 + threadIdx.x;
  if (p >= K * N) return;
  int KT   = K >> 5;
  int i    = p & 15;
  int lane = (p >> 4) & 31;
  int kt   = (p >> 9) % KT;
  int nt   = (p >> 9) / KT;
  int k = kt*32 + ((lane >> 4) << 3) + (i & 7) + ((i >> 3) << 4);
  int n = nt*16 + (lane & 15);
  P[p] = (bf16)W[(size_t)k * N + n];
}

__global__ void zero_kernel(float* __restrict__ p, int n) {
  int i = blockIdx.x * 256 + threadIdx.x;
  if (i < n) p[i] = 0.0f;
}

__global__ void copy_kernel(const float* __restrict__ s, float* __restrict__ d, int n) {
  int i = blockIdx.x * 256 + threadIdx.x;
  if (i < n) d[i] = s[i];
}

// ---------------------------------------------------------------------------
// h = x @ node_w + node_b   (M=50000, K=128, N=256) — no ReLU; mirrors to bf16
// ---------------------------------------------------------------------------
__global__ void __launch_bounds__(128)
node_proj_kernel(const float* __restrict__ x, const float* __restrict__ nb,
                 const bf16* __restrict__ Wp, float* __restrict__ h,
                 bf16* __restrict__ h_bf) {
  __shared__ __align__(16) bf16 sA[16 * NODE_IN];
  const int m0  = blockIdx.x * 16;
  const int tid = threadIdx.x;
  for (int idx = tid; idx < 16 * NODE_IN; idx += 128)
    sA[idx] = (bf16)x[(size_t)m0 * NODE_IN + idx];
  __syncthreads();

  const int lane = tid & 31, wid = tid >> 5;
  const int row = lane & 15, khalf = lane >> 4, mb = khalf * 8;
  const bf16* arow = sA + row * NODE_IN;
  v8f acc[4] = {};
  for (int kt = 0; kt < NODE_IN/32; ++kt) {
    v16bf a = load_a_tile(arow, kt*32, khalf);
#pragma unroll
    for (int j = 0; j < 4; ++j) {
      v16bf bm = load_b_tile(Wp, NODE_IN/32, wid*4 + j, kt, lane);
      acc[j] = wmma_bf16(a, bm, acc[j]);
    }
  }
#pragma unroll
  for (int j = 0; j < 4; ++j) {
    int col = (wid*4 + j)*16 + (lane & 15);
    float bias = nb[col];
#pragma unroll
    for (int v = 0; v < 8; ++v) {
      float val = acc[j][v] + bias;
      size_t o = (size_t)(m0 + mb + v) * HIDDEN + col;
      h[o] = val;
      h_bf[o] = (bf16)val;
    }
  }
}

// ---------------------------------------------------------------------------
// e = edge_attr @ edge_w + edge_b  (M=400000, K=32, N=32) — stored bf16
// ---------------------------------------------------------------------------
__global__ void __launch_bounds__(128)
edge_proj_kernel(const float* __restrict__ ea, const float* __restrict__ eb,
                 const bf16* __restrict__ Wp, bf16* __restrict__ e_out) {
  __shared__ __align__(16) bf16 sA[64 * EDGE_DIM];
  const int e0  = blockIdx.x * 64;
  const int tid = threadIdx.x;
  for (int idx = tid; idx < 64 * EDGE_DIM; idx += 128)
    sA[idx] = (bf16)ea[(size_t)e0 * EDGE_DIM + idx];
  __syncthreads();

  const int lane = tid & 31, wid = tid >> 5;
  const int khalf = lane >> 4, mb = khalf * 8;
  const bf16* arow = sA + ((lane & 15) + wid*16) * EDGE_DIM;
  v16bf a = load_a_tile(arow, 0, khalf);
  v8f acc[2] = {};
#pragma unroll
  for (int j = 0; j < 2; ++j) {
    v16bf bm = load_b_tile(Wp, 1, j, 0, lane);
    acc[j] = wmma_bf16(a, bm, acc[j]);
  }
#pragma unroll
  for (int j = 0; j < 2; ++j) {
    int col = j*16 + (lane & 15);
    float bias = eb[col];
#pragma unroll
    for (int v = 0; v < 8; ++v)
      e_out[(size_t)(e0 + wid*16 + mb + v) * EDGE_DIM + col] = (bf16)(acc[j][v] + bias);
  }
}

// ---------------------------------------------------------------------------
// Fused edge message, 64 edges/block, 256 threads (8 waves).
//   inp = [h[dst] | h[src] | e]  (K=544, bf16, async-gathered into LDS)
//   m = relu(relu(inp@W1+b1)@W2+b2);  aggr[dst] += m  (f32 atomics)
// Wave w computes 4 M-tiles x 2 N-tiles (nt = 2w, 2w+1); B reused x4, A x2.
// Dynamic LDS: [ sIdx(512B) | sA 64x544 bf16 (reused as sM 64x256 bf16) ]
// ---------------------------------------------------------------------------
__global__ void __launch_bounds__(256)
msg_kernel(const bf16* __restrict__ h_bf, const bf16* __restrict__ e_bf,
           const int* __restrict__ src, const int* __restrict__ dst,
           const float* __restrict__ b1, const float* __restrict__ b2,
           const bf16* __restrict__ W1p, const bf16* __restrict__ W2p,
           float* __restrict__ aggr) {
  extern __shared__ __align__(16) char smem[];
  int*  sIdx = (int*)smem;            // [0..63]=dst, [64..127]=src
  bf16* sA   = (bf16*)(smem + 512);   // 64 rows x 544
  bf16* sM   = (bf16*)(smem + 512);   // aliases sA after barrier (64 x 256)

  const int e0  = blockIdx.x * 64;
  const int tid = threadIdx.x;
  if (tid < 128) sIdx[tid] = (tid < 64) ? dst[e0 + tid] : src[e0 + tid - 64];
  __syncthreads();

  // --- async gather: h rows (64 rows x 2 halves x 512B) ----------------------
  const unsigned sA_off = lds_offset_of(sA);
  for (int c = tid; c < 4096; c += 256) {           // 16 chunks / thread
    int r = c >> 6, q = c & 63;
    int half = q >> 5, part = q & 31;               // 32 x 16B per 512B row
    int node = sIdx[r + 64*half];
    async_copy_b128(sA_off + (unsigned)(r*(MSG_K*2) + half*512 + part*16),
                    (const char*)h_bf + (size_t)node*512 + part*16);
  }
  // --- async gather: e rows (64 rows x 64B) ---------------------------------
  {
    int c = tid;                                    // 256 chunks total
    int r = c >> 2, part = c & 3;
    async_copy_b128(sA_off + (unsigned)(r*(MSG_K*2) + 1024 + part*16),
                    (const char*)e_bf + (size_t)(e0 + r)*64 + part*16);
  }
  async_wait0();
  __syncthreads();

  const int lane = tid & 31, wid = tid >> 5;
  const int row = lane & 15, khalf = lane >> 4, mb = khalf * 8;

  // --- layer 1: K=544 (17 tiles), N=256; 4Mx2N per wave ---------------------
  v8f acc[4][2] = {};
  {
    const bf16* arow0 = sA + row * MSG_K;
    for (int kt = 0; kt < 17; ++kt) {
      v16bf a[4];
#pragma unroll
      for (int mt = 0; mt < 4; ++mt)
        a[mt] = load_a_tile(arow0 + mt*16*MSG_K, kt*32, khalf);
#pragma unroll
      for (int j = 0; j < 2; ++j) {
        v16bf bm = load_b_tile(W1p, 17, wid*2 + j, kt, lane);
#pragma unroll
        for (int mt = 0; mt < 4; ++mt)
          acc[mt][j] = wmma_bf16(a[mt], bm, acc[mt][j]);
      }
    }
  }
  __syncthreads();   // all waves done reading sA; safe to overwrite with sM
#pragma unroll
  for (int j = 0; j < 2; ++j) {
    int col = (wid*2 + j)*16 + (lane & 15);
    float bias = b1[col];
#pragma unroll
    for (int mt = 0; mt < 4; ++mt)
#pragma unroll
      for (int v = 0; v < 8; ++v) {
        float val = acc[mt][j][v] + bias;
        sM[(mt*16 + mb + v) * HIDDEN + col] = (bf16)(val > 0.0f ? val : 0.0f);
      }
  }
  __syncthreads();

  // --- layer 2: K=256 (8 tiles), then scatter-add ---------------------------
  v8f acc2[4][2] = {};
  {
    const bf16* arow0 = sM + row * HIDDEN;
    for (int kt = 0; kt < 8; ++kt) {
      v16bf a[4];
#pragma unroll
      for (int mt = 0; mt < 4; ++mt)
        a[mt] = load_a_tile(arow0 + mt*16*HIDDEN, kt*32, khalf);
#pragma unroll
      for (int j = 0; j < 2; ++j) {
        v16bf bm = load_b_tile(W2p, 8, wid*2 + j, kt, lane);
#pragma unroll
        for (int mt = 0; mt < 4; ++mt)
          acc2[mt][j] = wmma_bf16(a[mt], bm, acc2[mt][j]);
      }
    }
  }
#pragma unroll
  for (int j = 0; j < 2; ++j) {
    int col = (wid*2 + j)*16 + (lane & 15);
    float bias = b2[col];
#pragma unroll
    for (int mt = 0; mt < 4; ++mt)
#pragma unroll
      for (int v = 0; v < 8; ++v) {
        float val = acc2[mt][j][v] + bias;
        val = val > 0.0f ? val : 0.0f;
        int node = sIdx[mt*16 + mb + v];
        unsafeAtomicAdd(&aggr[(size_t)node * HIDDEN + col], val);
      }
  }
}

// ---------------------------------------------------------------------------
// Node update: ui = [h | aggr] (K=512); h = relu(relu(ui@W1+b1)@W2+b2)
// In-place on h; also refreshes the bf16 mirror.
// ---------------------------------------------------------------------------
__global__ void __launch_bounds__(128)
update_kernel(float* __restrict__ h, bf16* __restrict__ h_bf,
              const float* __restrict__ aggr,
              const float* __restrict__ b1, const float* __restrict__ b2,
              const bf16* __restrict__ W1p, const bf16* __restrict__ W2p) {
  __shared__ __align__(16) bf16 sA[16 * 512];
  __shared__ __align__(16) bf16 sM[16 * HIDDEN];
  const int m0  = blockIdx.x * 16;
  const int tid = threadIdx.x;
  for (int idx = tid; idx < 16 * 512; idx += 128) {
    int r = idx >> 9, j = idx & 511;
    float v = (j < 256) ? h[(size_t)(m0 + r) * HIDDEN + j]
                        : aggr[(size_t)(m0 + r) * HIDDEN + (j - 256)];
    sA[idx] = (bf16)v;
  }
  __syncthreads();

  const int lane = tid & 31, wid = tid >> 5;
  const int row = lane & 15, khalf = lane >> 4, mb = khalf * 8;

  { // layer 1: K=512 (16 tiles)
    const bf16* arow = sA + row * 512;
    v8f acc[4] = {};
    for (int kt = 0; kt < 16; ++kt) {
      v16bf a = load_a_tile(arow, kt*32, khalf);
#pragma unroll
      for (int j = 0; j < 4; ++j) {
        v16bf bm = load_b_tile(W1p, 16, wid*4 + j, kt, lane);
        acc[j] = wmma_bf16(a, bm, acc[j]);
      }
    }
#pragma unroll
    for (int j = 0; j < 4; ++j) {
      int col = (wid*4 + j)*16 + (lane & 15);
      float bias = b1[col];
#pragma unroll
      for (int v = 0; v < 8; ++v) {
        float val = acc[j][v] + bias;
        sM[(mb + v) * HIDDEN + col] = (bf16)(val > 0.0f ? val : 0.0f);
      }
    }
  }
  __syncthreads();

  { // layer 2: K=256 (8 tiles) -> h, h_bf
    const bf16* arow = sM + row * HIDDEN;
    v8f acc[4] = {};
    for (int kt = 0; kt < 8; ++kt) {
      v16bf a = load_a_tile(arow, kt*32, khalf);
#pragma unroll
      for (int j = 0; j < 4; ++j) {
        v16bf bm = load_b_tile(W2p, 8, wid*4 + j, kt, lane);
        acc[j] = wmma_bf16(a, bm, acc[j]);
      }
    }
#pragma unroll
    for (int j = 0; j < 4; ++j) {
      int col = (wid*4 + j)*16 + (lane & 15);
      float bias = b2[col];
#pragma unroll
      for (int v = 0; v < 8; ++v) {
        float val = acc[j][v] + bias;
        val = val > 0.0f ? val : 0.0f;
        size_t o = (size_t)(m0 + mb + v) * HIDDEN + col;
        h[o] = val;
        h_bf[o] = (bf16)val;
      }
    }
  }
}

// ---------------------------------------------------------------------------
// Token head: t = relu(h@tok_w1+b1) [K=256, N=128]; logits = t@tok_w2+b2
// ---------------------------------------------------------------------------
__global__ void __launch_bounds__(128)
token_kernel(const bf16* __restrict__ h_bf, const float* __restrict__ b1,
             const bf16* __restrict__ W1p, const float* __restrict__ w2,
             const float* __restrict__ b2, float* __restrict__ out) {
  __shared__ __align__(16) bf16  sA[16 * HIDDEN];
  __shared__ float sT[16 * 128];
  const int m0  = blockIdx.x * 16;
  const int tid = threadIdx.x;
  for (int c = tid; c < 512; c += 128)   // 16 rows x 512B, 16B chunks
    *(v8bf*)(sA + c*8) = *(const v8bf*)(h_bf + (size_t)m0 * HIDDEN + c*8);
  __syncthreads();

  const int lane = tid & 31, wid = tid >> 5;
  const int row = lane & 15, khalf = lane >> 4, mb = khalf * 8;
  const bf16* arow = sA + row * HIDDEN;
  v8f acc[2] = {};
  for (int kt = 0; kt < 8; ++kt) {
    v16bf a = load_a_tile(arow, kt*32, khalf);
#pragma unroll
    for (int j = 0; j < 2; ++j) {
      v16bf bm = load_b_tile(W1p, 8, wid*2 + j, kt, lane);
      acc[j] = wmma_bf16(a, bm, acc[j]);
    }
  }
#pragma unroll
  for (int j = 0; j < 2; ++j) {
    int col = (wid*2 + j)*16 + (lane & 15);
    float bias = b1[col];
#pragma unroll
    for (int v = 0; v < 8; ++v) {
      float val = acc[j][v] + bias;
      sT[(mb + v) * 128 + col] = (val > 0.0f ? val : 0.0f);
    }
  }
  __syncthreads();
  if (tid < 16) {
    float s = 0.0f;
    for (int i = 0; i < 128; ++i) s += sT[tid * 128 + i] * w2[i];
    out[m0 + tid] = s + b2[0];
  }
}

// ---------------------------------------------------------------------------
extern "C" void kernel_launch(void* const* d_in, const int* in_sizes, int n_in,
                              void* d_out, int out_size, void* d_ws, size_t ws_size,
                              hipStream_t stream) {
  const float* x      = (const float*)d_in[0];
  const int*   ei     = (const int*)  d_in[1];
  const float* ea     = (const float*)d_in[2];
  const float* node_w = (const float*)d_in[3];
  const float* node_b = (const float*)d_in[4];
  const float* edge_w = (const float*)d_in[5];
  const float* edge_b = (const float*)d_in[6];
  const float* msg_w1 = (const float*)d_in[7];
  const float* msg_b1 = (const float*)d_in[8];
  const float* msg_w2 = (const float*)d_in[9];
  const float* msg_b2 = (const float*)d_in[10];
  const float* up_w1  = (const float*)d_in[11];
  const float* up_b1  = (const float*)d_in[12];
  const float* up_w2  = (const float*)d_in[13];
  const float* up_b2  = (const float*)d_in[14];
  const float* tok_w1 = (const float*)d_in[15];
  const float* tok_b1 = (const float*)d_in[16];
  const float* tok_w2 = (const float*)d_in[17];
  const float* tok_b2 = (const float*)d_in[18];
  (void)in_sizes; (void)n_in; (void)out_size; (void)ws_size;

  const int* src = ei;             // edge_index[0]
  const int* dst = ei + N_EDGES;   // edge_index[1]

  // workspace carving (256B-aligned)
  char*  ws  = (char*)d_ws;
  size_t off = 0;
  auto carve = [&](size_t bytes) -> void* {
    void* p = ws + off;
    off = (off + bytes + 255) & ~(size_t)255;
    return p;
  };
  float* h      = (float*)carve((size_t)N_NODES * HIDDEN * 4);
  bf16*  h_bf   = (bf16*) carve((size_t)N_NODES * HIDDEN * 2);
  float* aggr   = (float*)carve((size_t)N_NODES * HIDDEN * 4);
  bf16*  e_bf   = (bf16*) carve((size_t)N_EDGES * EDGE_DIM * 2);
  bf16*  p_node = (bf16*) carve((size_t)NODE_IN * HIDDEN * 2);
  bf16*  p_edge = (bf16*) carve((size_t)EDGE_DIM * EDGE_DIM * 2);
  bf16*  p_mw1  = (bf16*) carve((size_t)3 * MSG_K * HIDDEN * 2);
  bf16*  p_mw2  = (bf16*) carve((size_t)3 * HIDDEN * HIDDEN * 2);
  bf16*  p_uw1  = (bf16*) carve((size_t)3 * 512 * HIDDEN * 2);
  bf16*  p_uw2  = (bf16*) carve((size_t)3 * HIDDEN * HIDDEN * 2);
  bf16*  p_tw1  = (bf16*) carve((size_t)HIDDEN * 128 * 2);

  auto pack = [&](const float* W, bf16* P, int K, int N) {
    int total = K * N;
    pack_w_kernel<<<(total + 255) / 256, 256, 0, stream>>>(W, P, K, N);
  };
  pack(node_w, p_node, NODE_IN, HIDDEN);
  pack(edge_w, p_edge, EDGE_DIM, EDGE_DIM);
  for (int l = 0; l < 3; ++l) {
    pack(msg_w1 + (size_t)l * MSG_K * HIDDEN,  p_mw1 + (size_t)l * MSG_K * HIDDEN,  MSG_K,  HIDDEN);
    pack(msg_w2 + (size_t)l * HIDDEN * HIDDEN, p_mw2 + (size_t)l * HIDDEN * HIDDEN, HIDDEN, HIDDEN);
    pack(up_w1  + (size_t)l * 512 * HIDDEN,    p_uw1 + (size_t)l * 512 * HIDDEN,    512,    HIDDEN);
    pack(up_w2  + (size_t)l * HIDDEN * HIDDEN, p_uw2 + (size_t)l * HIDDEN * HIDDEN, HIDDEN, HIDDEN);
  }
  pack(tok_w1, p_tw1, HIDDEN, 128);

  node_proj_kernel<<<N_NODES / 16, 128, 0, stream>>>(x, node_b, p_node, h, h_bf);
  edge_proj_kernel<<<N_EDGES / 64, 128, 0, stream>>>(ea, edge_b, p_edge, e_bf);

  const int nh = N_NODES * HIDDEN;
  const size_t msg_lds = 512 + (size_t)64 * MSG_K * 2;   // 70,144 B dynamic LDS
  for (int l = 0; l < 3; ++l) {
    zero_kernel<<<(nh + 255) / 256, 256, 0, stream>>>(aggr, nh);
    msg_kernel<<<N_EDGES / 64, 256, msg_lds, stream>>>(
        h_bf, e_bf, src, dst,
        msg_b1 + (size_t)l * HIDDEN, msg_b2 + (size_t)l * HIDDEN,
        p_mw1 + (size_t)l * MSG_K * HIDDEN, p_mw2 + (size_t)l * HIDDEN * HIDDEN,
        aggr);
    update_kernel<<<N_NODES / 16, 128, 0, stream>>>(
        h, h_bf, aggr,
        up_b1 + (size_t)l * HIDDEN, up_b2 + (size_t)l * HIDDEN,
        p_uw1 + (size_t)l * 512 * HIDDEN, p_uw2 + (size_t)l * HIDDEN * HIDDEN);
  }

  float* out = (float*)d_out;
  token_kernel<<<N_NODES / 16, 128, 0, stream>>>(h_bf, tok_b1, p_tw1, tok_w2, tok_b2, out);
  copy_kernel<<<(nh + 255) / 256, 256, 0, stream>>>(h, out + N_NODES, nh);
}
